// model_63471026700329
// MI455X (gfx1250) — compile-verified, matrix operand order checked by reading, not run on hardware
//
#include <hip/hip_runtime.h>

// ---------------------------------------------------------------------------
// Fused persistent GRU seq2seq for MI455X (gfx1250, wave32, WMMA).
//
//  * encoder_kernel: 3 stacked GRU layers fused per timestep, 512 steps.
//  * decoder_kernel: 3 GRUCells + linear head, 512 steps.
//  * All weights staged into LDS once (f16, 144B conflict-free row stride);
//    the 1024-step recurrent chain runs out of LDS + registers.
//  * WMMA accumulators start at inline-0 (no bias splat on the critical
//    path); all biases are folded into the gate pass via small combined
//    LDS arrays (wave-broadcast reads).
// ---------------------------------------------------------------------------

#define H       60          // hidden size
#define G3      180         // 3*H gate dim
#define NP      192         // padded gate dim (12 x 16 tiles)
#define KP      64          // padded K (hidden) dim, global packed stride
#define KSTR    72          // LDS weight row stride in halves (144B)
#define BT      32          // batch rows per workgroup
#define HSTR    72          // LDS h row stride in halves (144B)
#define SSTR    36          // LDS S/G row stride in floats (144B)
#define SEQ     512
#define BATCH   2048
#define THREADS 256

typedef _Float16 v8h  __attribute__((ext_vector_type(8)));
typedef _Float16 v16h __attribute__((ext_vector_type(16)));
typedef float    v8f  __attribute__((ext_vector_type(8)));
typedef float    v4f  __attribute__((ext_vector_type(4)));

// ---- workspace layout (bytes) ---------------------------------------------
#define OFF_WHHT   0                         // [NP][KP] f16
#define OFF_WIHT   24576                     // [NP][KP] f16 (din==H cells)
#define OFF_BIH    49152                     // [NP] f32
#define OFF_BHH    49920                     // [NP] f32
#define OFF_WIHV   50688                     // [NP] f32 (din==1 cells)
#define CELL_STRIDE 51456
#define OFF_LIN    (6 * CELL_STRIDE)         // 128 floats: lw[0..63], lb at [64]
#define OFF_HT1    (OFF_LIN + 512)
#define HT_BYTES   (BATCH * H * 4)
#define OFF_HT2    (OFF_HT1 + HT_BYTES)
#define OFF_HT3    (OFF_HT2 + HT_BYTES)

// LDS-resident cell weights / biases
struct CellW {
  const _Float16* whhT;   // [NP][KSTR]
  const _Float16* wihT;   // [NP][KSTR] or nullptr
  const float*    wihv;   // [NP] or nullptr (din==1)
  const float*    brz;    // [120]  bih+bhh for r/z gate columns
  const float*    bin;    // [60]   bih for n gate
  const float*    bhn;    // [60]   bhh for n gate
};

// 16-bit WMMA operand fragment: lane holds {k..k+7} and {k+16..k+23} of one
// row/column -> two contiguous 16B loads.
__device__ __forceinline__ v16h ldfrag(const _Float16* p) {
  v8h lo = *(const v8h*)p;
  v8h hi = *(const v8h*)(p + 16);
  return __builtin_shufflevector(lo, hi, 0,1,2,3,4,5,6,7,8,9,10,11,12,13,14,15);
}

__device__ __forceinline__ v8f wmma32(v16h a, v16h b, v8f c) {
  return __builtin_amdgcn_wmma_f32_16x16x32_f16(
      /*neg_a=*/false, a, /*neg_b=*/false, b,
      /*c_mod=*/(short)0, c, /*reuse_a=*/false, /*reuse_b=*/false);
}

__device__ __forceinline__ float sigf(float x)  { return 1.f / (1.f + __expf(-x)); }
__device__ __forceinline__ float tanhf_(float x){ float e = __expf(-2.f * x); return (1.f - e) / (1.f + e); }

// Copy one packed f16 weight matrix [NP][KP] (global) -> [NP][KSTR] (LDS).
__device__ __forceinline__ void copy_w16(const _Float16* __restrict__ g,
                                         _Float16* __restrict__ l, int tid) {
  const unsigned* gs = (const unsigned*)g;
  unsigned* ls = (unsigned*)l;
  for (int i = tid; i < NP * (KP / 2); i += THREADS) {
    const int row = i >> 5, d = i & 31;           // 32 dwords per row
    ls[row * (KSTR / 2) + d] = gs[row * (KP / 2) + d];
  }
}

__device__ __forceinline__ void copy_f32(const float* __restrict__ g,
                                         float* __restrict__ l, int n, int tid) {
  for (int i = tid; i < n; i += THREADS) l[i] = g[i];
}

// Build combined gate-bias arrays in LDS from the packed f32 biases.
__device__ __forceinline__ void stage_bias(const float* __restrict__ bih,
                                           const float* __restrict__ bhh,
                                           float* __restrict__ brz,
                                           float* __restrict__ bin,
                                           float* __restrict__ bhn, int tid) {
  for (int i = tid; i < 2 * H; i += THREADS) brz[i] = bih[i] + bhh[i];
  for (int i = tid; i < H; i += THREADS) { bin[i] = bih[2 * H + i]; bhn[i] = bhh[2 * H + i]; }
}

// One GRU cell step for a [BT][H] state tile resident in LDS.
//   GEMM_IN=true : input is previous layer's h tile (LDS f16, K=64 GEMM)
//   GEMM_IN=false: input is a scalar per batch row (rank-1 update)
template <bool GEMM_IN>
__device__ void gru_step(_Float16* hcell, const _Float16* xin, const float* xbuf,
                         float* S, float* G, const CellW w, int tid)
{
  __syncthreads();  // prior gate-pass / staging writes visible

  const int wv    = __builtin_amdgcn_readfirstlane(tid) >> 5;  // scalar wave id
  const int lane  = tid & 31;
  const int m0    = (wv & 1) << 4;     // 0 or 16 (scalar)
  const int third = wv >> 1;           // 0..3 -> n-tiles [third*3, third*3+3)
  const int ln    = lane & 15;
  const int lk    = (lane >> 4) << 3;  // 0 or 8
  const int arow  = m0 + ln;
  const int mb    = m0 + lk;           // 0,8,16,24

  // A fragments: recurrent state rows (K cols >= 60 are zero pad)
  v16h ah0 = ldfrag(hcell + arow * HSTR + lk);
  v16h ah1 = ldfrag(hcell + arow * HSTR + 32 + lk);
  v16h ax0{}, ax1{};
  v4f  xlo{}, xhi{};
  if constexpr (GEMM_IN) {
    ax0 = ldfrag(xin + arow * HSTR + lk);
    ax1 = ldfrag(xin + arow * HSTR + 32 + lk);
  } else {
    xlo = *(const v4f*)(xbuf + mb);
    xhi = *(const v4f*)(xbuf + mb + 4);
  }

#pragma unroll
  for (int i = 0; i < 3; ++i) {
    const int nt = third * 3 + i;      // scalar tile index 0..11
    const int n  = (nt << 4) + ln;

    // gh = h @ whh^T, accumulator starts at inline 0 (no bias splat)
    v8f c = {};
    c = wmma32(ah0, ldfrag(w.whhT + n * KSTR + lk),      c);
    c = wmma32(ah1, ldfrag(w.whhT + n * KSTR + 32 + lk), c);

    // spill pure gh for the n-gate columns (cols 112..191 cover 120..179);
    // scalar predicate -> scalar branch
    if (nt >= 7) {
      float* gp = G + (n - 112) * SSTR + mb;
      *(v4f*)gp       = __builtin_shufflevector(c, c, 0, 1, 2, 3);
      *(v4f*)(gp + 4) = __builtin_shufflevector(c, c, 4, 5, 6, 7);
    }

    // gi contribution -> C = gi + gh (biases handled in gate pass)
    if constexpr (GEMM_IN) {
      c = wmma32(ax0, ldfrag(w.wihT + n * KSTR + lk),      c);
      c = wmma32(ax1, ldfrag(w.wihT + n * KSTR + 32 + lk), c);
    } else {
      const float wvv = w.wihv[n];
#pragma unroll
      for (int r = 0; r < 4; ++r) { c[r] += xlo[r] * wvv; c[4 + r] += xhi[r] * wvv; }
    }

    // store transposed S[n][m]: D-fragment rows contiguous in m per lane
    float* sp = S + n * SSTR + mb;
    *(v4f*)sp       = __builtin_shufflevector(c, c, 0, 1, 2, 3);
    *(v4f*)(sp + 4) = __builtin_shufflevector(c, c, 4, 5, 6, 7);
  }

  __syncthreads();  // S/G complete

  // gate nonlinearities + state update (h stays f16 in LDS).
  //   S_r = i_r+h_r (no bias), S_z likewise, S_n = i_n+gh_n, G = gh_n
  //   r = sig(S_r + brz[j]);  z = sig(S_z + brz[60+j])
  //   n = tanh(i_n + bin + r*(gh_n + bhn)) = tanh(S_n + bin - G + r*(G+bhn))
  for (int e = tid; e < BT * H; e += THREADS) {
    const int b = e & (BT - 1);
    const int j = e >> 5;  // BT == 32
    const float r   = sigf(S[j * SSTR + b] + w.brz[j]);
    const float z   = sigf(S[(H + j) * SSTR + b] + w.brz[H + j]);
    const float g   = G[(8 + j) * SSTR + b];                      // gh_n (col 120+j)
    const float nn  = tanhf_(S[(2 * H + j) * SSTR + b] + w.bin[j] - g + r * (g + w.bhn[j]));
    const float ho  = (float)hcell[b * HSTR + j];
    hcell[b * HSTR + j] = (_Float16)((1.f - z) * nn + z * ho);
  }
}

// ---------------------------------------------------------------------------
__global__ __launch_bounds__(THREADS)
void encoder_kernel(const float* __restrict__ inputs, char* __restrict__ ws)
{
  // LDS-resident weights for the 3 encoder cells (~138 KB) + working set.
  __shared__ __align__(16) _Float16 lw_whh[3][NP * KSTR];
  __shared__ __align__(16) _Float16 lw_wih[2][NP * KSTR];   // cells 1,2
  __shared__ __align__(16) float lw_brz[3][2 * H];
  __shared__ __align__(16) float lw_bin[3][H];
  __shared__ __align__(16) float lw_bhn[3][H];
  __shared__ __align__(16) float lw_wihv[NP];
  __shared__ __align__(16) _Float16 h0s[BT * HSTR], h1s[BT * HSTR], h2s[BT * HSTR];
  __shared__ __align__(16) float S[NP * SSTR];
  __shared__ __align__(16) float G[80 * SSTR];
  __shared__ __align__(16) float xb[BT];

  const int tid   = threadIdx.x;
  const int bbase = blockIdx.x * BT;

  // stage weights global -> LDS (one time)
  for (int c = 0; c < 3; ++c) {
    const char* base = ws + (size_t)c * CELL_STRIDE;
    copy_w16((const _Float16*)(base + OFF_WHHT), lw_whh[c], tid);
    if (c > 0) copy_w16((const _Float16*)(base + OFF_WIHT), lw_wih[c - 1], tid);
    stage_bias((const float*)(base + OFF_BIH), (const float*)(base + OFF_BHH),
               lw_brz[c], lw_bin[c], lw_bhn[c], tid);
  }
  copy_f32((const float*)(ws + OFF_WIHV), lw_wihv, NP, tid);

  const CellW w0 = { lw_whh[0], nullptr,   lw_wihv, lw_brz[0], lw_bin[0], lw_bhn[0] };
  const CellW w1 = { lw_whh[1], lw_wih[0], nullptr, lw_brz[1], lw_bin[1], lw_bhn[1] };
  const CellW w2 = { lw_whh[2], lw_wih[1], nullptr, lw_brz[2], lw_bin[2], lw_bhn[2] };

  for (int i = tid; i < BT * HSTR; i += THREADS) {
    h0s[i] = (_Float16)0.f; h1s[i] = (_Float16)0.f; h2s[i] = (_Float16)0.f;
  }

  for (int t = 0; t < SEQ; ++t) {
    if (tid < BT) xb[tid] = inputs[(size_t)(bbase + tid) * SEQ + t];
    gru_step<false>(h0s, nullptr, xb,      S, G, w0, tid);
    gru_step<true >(h1s, h0s,     nullptr, S, G, w1, tid);
    gru_step<true >(h2s, h1s,     nullptr, S, G, w2, tid);
  }

  __syncthreads();
  float* ht1 = (float*)(ws + OFF_HT1);
  float* ht2 = (float*)(ws + OFF_HT2);
  float* ht3 = (float*)(ws + OFF_HT3);
  for (int e = tid; e < BT * H; e += THREADS) {
    const int b = e & (BT - 1), j = e >> 5;
    ht1[(size_t)(bbase + b) * H + j] = (float)h2s[b * HSTR + j];  // top layer -> cell1
    ht2[(size_t)(bbase + b) * H + j] = (float)h1s[b * HSTR + j];  // mid layer -> cell2
    ht3[(size_t)(bbase + b) * H + j] = (float)h0s[b * HSTR + j];  // bottom    -> cell3
  }
}

__global__ __launch_bounds__(THREADS)
void decoder_kernel(const float* __restrict__ outputs, char* __restrict__ ws,
                    float* __restrict__ out)
{
  __shared__ __align__(16) _Float16 lw_whh[3][NP * KSTR];
  __shared__ __align__(16) _Float16 lw_wih[2][NP * KSTR];   // cells c2,c3
  __shared__ __align__(16) float lw_brz[3][2 * H];
  __shared__ __align__(16) float lw_bin[3][H];
  __shared__ __align__(16) float lw_bhn[3][H];
  __shared__ __align__(16) float lw_wihv[NP];
  __shared__ __align__(16) _Float16 h1s[BT * HSTR], h2s[BT * HSTR], h3s[BT * HSTR];
  __shared__ __align__(16) float S[NP * SSTR];
  __shared__ __align__(16) float G[80 * SSTR];
  __shared__ __align__(16) float xb[BT];
  __shared__ __align__(16) float lw[64];
  __shared__ float lb;

  const int tid   = threadIdx.x;
  const int bbase = blockIdx.x * BT;

  for (int c = 0; c < 3; ++c) {
    const char* base = ws + (size_t)(3 + c) * CELL_STRIDE;
    copy_w16((const _Float16*)(base + OFF_WHHT), lw_whh[c], tid);
    if (c > 0) copy_w16((const _Float16*)(base + OFF_WIHT), lw_wih[c - 1], tid);
    stage_bias((const float*)(base + OFF_BIH), (const float*)(base + OFF_BHH),
               lw_brz[c], lw_bin[c], lw_bhn[c], tid);
  }
  copy_f32((const float*)(ws + 3 * CELL_STRIDE + OFF_WIHV), lw_wihv, NP, tid);

  const CellW w1 = { lw_whh[0], nullptr,   lw_wihv, lw_brz[0], lw_bin[0], lw_bhn[0] };
  const CellW w2 = { lw_whh[1], lw_wih[0], nullptr, lw_brz[1], lw_bin[1], lw_bhn[1] };
  const CellW w3 = { lw_whh[2], lw_wih[1], nullptr, lw_brz[2], lw_bin[2], lw_bhn[2] };

  const float* ht1 = (const float*)(ws + OFF_HT1);
  const float* ht2 = (const float*)(ws + OFF_HT2);
  const float* ht3 = (const float*)(ws + OFF_HT3);
  const float* lin = (const float*)(ws + OFF_LIN);

  for (int e = tid; e < BT * HSTR; e += THREADS) {
    const int b = e / HSTR, j = e % HSTR;
    const size_t gi = (size_t)(bbase + b) * H + j;
    h1s[e] = (j < H) ? (_Float16)ht1[gi] : (_Float16)0.f;
    h2s[e] = (j < H) ? (_Float16)ht2[gi] : (_Float16)0.f;
    h3s[e] = (j < H) ? (_Float16)ht3[gi] : (_Float16)0.f;
  }
  if (tid < 64) lw[tid] = lin[tid];
  if (tid == 0) lb = lin[64];

  for (int t = 0; t < SEQ; ++t) {
    if (tid < BT) xb[tid] = outputs[(size_t)(bbase + tid) * SEQ + t];
    gru_step<false>(h1s, nullptr, xb,      S, G, w1, tid);
    gru_step<true >(h2s, h1s,     nullptr, S, G, w2, tid);
    gru_step<true >(h3s, h2s,     nullptr, S, G, w3, tid);
    __syncthreads();
    if (tid < BT) {
      float acc = lb;
#pragma unroll
      for (int j = 0; j < H; ++j) acc += (float)h3s[tid * HSTR + j] * lw[j];
      out[(size_t)(bbase + tid) * SEQ + t] = acc;
    }
  }
}

// ---------------------------------------------------------------------------
// Weight packing: f32 -> padded f16 [NP][KP] (B operand layout), biases, and
// rank-1 input columns for the din==1 cells.
__global__ void pack_cell(const float* __restrict__ wih, const float* __restrict__ whh,
                          const float* __restrict__ bih, const float* __restrict__ bhh,
                          int din, char* __restrict__ base)
{
  _Float16* whhT = (_Float16*)(base + OFF_WHHT);
  _Float16* wihT = (_Float16*)(base + OFF_WIHT);
  float* bihp = (float*)(base + OFF_BIH);
  float* bhhp = (float*)(base + OFF_BHH);
  float* wihv = (float*)(base + OFF_WIHV);

  const int idx = blockIdx.x * blockDim.x + threadIdx.x;
  if (idx >= NP * KP) return;
  const int n = idx >> 6, k = idx & 63;
  whhT[idx] = (n < G3 && k < H) ? (_Float16)whh[n * H + k] : (_Float16)0.f;
  _Float16 wvv = (_Float16)0.f;
  if (n < G3 && din > 1 && k < din) wvv = (_Float16)wih[n * din + k];
  wihT[idx] = wvv;
  if (k == 0) {
    bihp[n] = (n < G3) ? bih[n] : 0.f;
    bhhp[n] = (n < G3) ? bhh[n] : 0.f;
    wihv[n] = (n < G3 && din == 1) ? wih[n] : 0.f;
  }
}

__global__ void pack_lin(const float* __restrict__ lwi, const float* __restrict__ lbi,
                         float* __restrict__ lp)
{
  const int j = threadIdx.x;
  if (j < 64) lp[j] = (j < H) ? lwi[j] : 0.f;
  if (j == 0) lp[64] = lbi[0];
}

// ---------------------------------------------------------------------------
extern "C" void kernel_launch(void* const* d_in, const int* in_sizes, int n_in,
                              void* d_out, int out_size, void* d_ws, size_t ws_size,
                              hipStream_t stream)
{
  (void)in_sizes; (void)n_in; (void)out_size; (void)ws_size;
  const float* inputs  = (const float*)d_in[0];
  const float* outputs = (const float*)d_in[1];
  char* ws = (char*)d_ws;

  // cells: enc0, enc1, enc2, c1, c2, c3  (d_in[2..25], 4 tensors each)
  const int din[6] = {1, H, H, 1, H, H};
  for (int c = 0; c < 6; ++c) {
    const int p = 2 + c * 4;
    pack_cell<<<(NP * KP + 255) / 256, 256, 0, stream>>>(
        (const float*)d_in[p],     (const float*)d_in[p + 1],
        (const float*)d_in[p + 2], (const float*)d_in[p + 3],
        din[c], ws + (size_t)c * CELL_STRIDE);
  }
  pack_lin<<<1, 64, 0, stream>>>((const float*)d_in[26], (const float*)d_in[27],
                                 (float*)(ws + OFF_LIN));

  encoder_kernel<<<BATCH / BT, THREADS, 0, stream>>>(inputs, ws);
  decoder_kernel<<<BATCH / BT, THREADS, 0, stream>>>(outputs, ws, (float*)d_out);
}
